// SupconLoss_87548613362417
// MI455X (gfx1250) — matrix operand order repaired and checked by reading, not compiled, not run on hardware
//
#include <hip/hip_runtime.h>
#include <hip/hip_bf16.h>
#include <math.h>

typedef __attribute__((ext_vector_type(16))) _Float16 v16h;
typedef __attribute__((ext_vector_type(8)))  _Float16 v8h;
typedef __attribute__((ext_vector_type(8)))  float    v8f;

#define NROWS 8192
#define DDIM  128
#define EPSF  1e-8f
#define INV_TEMP (1.0f / 0.07f)
#define PROBF 0.1f

#define WAVES_PER_BLOCK 8
#define ROW_TILES (NROWS / 16)                // 512 row tiles of 16
#define JCHUNKS 16
#define TILES_PER_CHUNK (ROW_TILES / JCHUNKS) // 32 column tiles per wave
#define PF_DIST 4                             // prefetch distance (tiles)

union V16U { v16h v; v8h h[2]; };

struct BT { v16h t[4]; };

// Load one 32x16 B fragment set (4 K-slices) for this lane.
// base = xnh + (jt*16 + l16)*DDIM + hf*16  (lane-half K offset pre-applied)
__device__ __forceinline__ BT load_btile(const _Float16* __restrict__ base) {
    BT r;
    #pragma unroll
    for (int tk = 0; tk < 4; ++tk) {
        V16U u;
        u.h[0] = *(const v8h*)(base + tk * 32);
        u.h[1] = *(const v8h*)(base + tk * 32 + 8);
        r.t[tk] = u.v;
    }
    return r;
}

__device__ __forceinline__ v8f wmma4(const v16h A[4], const BT& b) {
    v8f c = {0.f, 0.f, 0.f, 0.f, 0.f, 0.f, 0.f, 0.f};
    #pragma unroll
    for (int tk = 0; tk < 4; ++tk)
        c = __builtin_amdgcn_wmma_f32_16x16x32_f16(
                false, A[tk], false, b.t[tk], (short)0, c, false, false);
    return c;
}

// ---------------- init: zero the f32 accumulator region ----------------
__global__ void zero_f32(float* p, int n) {
    int i = blockIdx.x * blockDim.x + threadIdx.x;
    if (i < n) p[i] = 0.0f;
}

// ---------------- normalize rows, write f16 xn ----------------
__global__ void __launch_bounds__(128) normalize_rows(const float* __restrict__ reps,
                                                      _Float16* __restrict__ xnh) {
    __shared__ float sred[4];
    const int row = blockIdx.x;
    const int t   = threadIdx.x;           // 0..127
    const int lane = t & 31, wv = t >> 5;
    float v = reps[(size_t)row * DDIM + t];
    float s = v * v;
    #pragma unroll
    for (int off = 1; off < 32; off <<= 1) s += __shfl_xor(s, off, 32);
    if (lane == 0) sred[wv] = s;
    __syncthreads();
    float tot = sred[0] + sred[1] + sred[2] + sred[3];
    float nrm = fmaxf(sqrtf(tot), EPSF);
    xnh[(size_t)row * DDIM + t] = (_Float16)(v / nrm);
}

// ---------------- pass A: global max of masked scores ----------------
__global__ void __launch_bounds__(256) pass_max(const _Float16* __restrict__ xnh,
                                                float* __restrict__ maxf) {
    const int lane = threadIdx.x & 31;
    const int wv   = threadIdx.x >> 5;
    const int hf   = lane >> 4;            // lane-half: 0 or 1
    const int l16  = lane & 15;
    const int I    = blockIdx.x * WAVES_PER_BLOCK + wv;   // row tile index

    // Preload A tile for all 4 K-slices (contiguous K = kk + 16*hf + s per lane-half, same as B)
    const _Float16* arow = xnh + (size_t)(I * 16 + l16) * DDIM;
    v16h A[4];
    #pragma unroll
    for (int tk = 0; tk < 4; ++tk) {
        const int kk = tk * 32;
        V16U ua;
        ua.h[0] = *(const v8h*)(arow + kk + hf * 16);      // K = kk + hf*16 .. +8
        ua.h[1] = *(const v8h*)(arow + kk + hf * 16 + 8);  // K = kk+hf*16+8 .. +8
        A[tk] = ua.v;
    }

    const int baseM = I * 16 + hf * 8;
    const int j0    = blockIdx.y * TILES_PER_CHUNK;
    const int jend  = j0 + TILES_PER_CHUNK;
    const _Float16* bbase = xnh + (size_t)l16 * DDIM + hf * 16;  // B = A^T, + jt*16*DDIM

    float vmax = 0.0f;
    BT cur = load_btile(bbase + (size_t)j0 * 16 * DDIM);
    for (int jt = j0; jt < jend; ++jt) {
        if (jt + PF_DIST < jend)
            __builtin_prefetch(bbase + (size_t)(jt + PF_DIST) * 16 * DDIM, 0, 0);
        const int jn = (jt + 1 < jend) ? jt + 1 : jt;
        BT nxt = load_btile(bbase + (size_t)jn * 16 * DDIM);   // issued before WMMA on cur

        v8f c = wmma4(A, cur);

        const int n_g = jt * 16 + l16;
        #pragma unroll
        for (int v = 0; v < 8; ++v) {
            float s = ((1.0f + c[v]) * 0.5f + EPSF) * INV_TEMP;
            s = (baseM + v == n_g) ? 0.0f : s;   // diagonal masked to 0
            vmax = fmaxf(vmax, s);
        }
        cur = nxt;
    }
    #pragma unroll
    for (int off = 1; off < 32; off <<= 1) vmax = fmaxf(vmax, __shfl_xor(vmax, off, 32));
    if (lane == 0)
        atomicMax((unsigned int*)maxf, __float_as_uint(vmax));  // positive floats: bit-monotonic
}

// ---------------- pass B: fused GEMM + masked exp row sums ----------------
__global__ void __launch_bounds__(256) pass_sums(const _Float16* __restrict__ xnh,
                                                 const int* __restrict__ labels,
                                                 const float* __restrict__ maxf,
                                                 float* __restrict__ pos_sum,
                                                 float* __restrict__ neg_sum,
                                                 float* __restrict__ cnt_sum) {
    const int lane = threadIdx.x & 31;
    const int wv   = threadIdx.x >> 5;
    const int hf   = lane >> 4;
    const int l16  = lane & 15;
    const int I    = blockIdx.x * WAVES_PER_BLOCK + wv;

    const float M = __uint_as_float(*(const unsigned int*)maxf);

    const _Float16* arow = xnh + (size_t)(I * 16 + l16) * DDIM;
    v16h A[4];
    #pragma unroll
    for (int tk = 0; tk < 4; ++tk) {
        const int kk = tk * 32;
        V16U ua;
        ua.h[0] = *(const v8h*)(arow + kk + hf * 16);
        ua.h[1] = *(const v8h*)(arow + kk + hf * 16 + 8);
        A[tk] = ua.v;
    }

    const int baseM = I * 16 + hf * 8;     // C-layout: lane-half picks M=0..7 vs 8..15
    int   labR[8];
    bool  nnR[8];
    #pragma unroll
    for (int v = 0; v < 8; ++v) { labR[v] = labels[baseM + v]; nnR[v] = (labR[v] != 0); }

    float posA[8], negA[8], cntA[8];
    #pragma unroll
    for (int v = 0; v < 8; ++v) { posA[v] = 0.f; negA[v] = 0.f; cntA[v] = 0.f; }

    const int j0   = blockIdx.y * TILES_PER_CHUNK;
    const int jend = j0 + TILES_PER_CHUNK;
    const _Float16* bbase = xnh + (size_t)l16 * DDIM + hf * 16;

    BT cur = load_btile(bbase + (size_t)j0 * 16 * DDIM);
    for (int jt = j0; jt < jend; ++jt) {
        if (jt + PF_DIST < jend)
            __builtin_prefetch(bbase + (size_t)(jt + PF_DIST) * 16 * DDIM, 0, 0);
        const int jn = (jt + 1 < jend) ? jt + 1 : jt;
        BT nxt = load_btile(bbase + (size_t)jn * 16 * DDIM);   // overlap with WMMA on cur

        v8f c = wmma4(A, cur);

        const int  n_g  = jt * 16 + l16;
        const int  labC = labels[n_g];
        const bool nnC  = (labC != 0);
        #pragma unroll
        for (int v = 0; v < 8; ++v) {
            const float s = ((1.0f + c[v]) * 0.5f + EPSF) * INV_TEMP - M;
            const float e = __expf(s);
            const bool diag = (baseM + v == n_g);
            const bool pos  = (labR[v] == labC);
            const bool neu  = (nnR[v] != nnC);       // exactly one of the pair is neutral
            posA[v] += (!diag && pos)  ? e : 0.0f;
            negA[v] += (!diag && !pos) ? e : 0.0f;
            negA[v] += (!diag && neu)  ? PROBF * e : 0.0f;
            cntA[v] += (!diag && pos)  ? 1.0f : 0.0f;
        }
        cur = nxt;
    }

    // Row r = baseM + v; its 16 column-partials live across the 16-lane half-group.
    #pragma unroll
    for (int v = 0; v < 8; ++v) {
        float p = posA[v], n = negA[v], k = cntA[v];
        #pragma unroll
        for (int off = 1; off < 16; off <<= 1) {
            p += __shfl_xor(p, off, 32);
            n += __shfl_xor(n, off, 32);
            k += __shfl_xor(k, off, 32);
        }
        if (l16 == 0) {
            const int r = baseM + v;
            atomicAdd(pos_sum + r, p);
            atomicAdd(neg_sum + r, n);
            atomicAdd(cnt_sum + r, k);
        }
    }
}

// ---------------- per-row loss + masked accumulation ----------------
__global__ void finalize_rows(const float* __restrict__ pos_sum,
                              const float* __restrict__ neg_sum,
                              const float* __restrict__ cnt_sum,
                              float* __restrict__ acc) {
    const int i = blockIdx.x * blockDim.x + threadIdx.x;
    if (i >= NROWS) return;
    const float p = pos_sum[i], n = neg_sum[i], k = cnt_sum[i];
    float probs = p / (p + n);
    probs = probs / (k + EPSF);
    const float loss = -__logf(probs + EPSF);
    const float m = (loss > 0.3f) ? 1.0f : 0.0f;
    atomicAdd(&acc[0], loss * m);
    atomicAdd(&acc[1], m);
}

__global__ void write_out(const float* __restrict__ acc, float* __restrict__ out) {
    out[0] = acc[0] / (acc[1] + EPSF);
}

// ---------------- host-side launch ----------------
extern "C" void kernel_launch(void* const* d_in, const int* in_sizes, int n_in,
                              void* d_out, int out_size, void* d_ws, size_t ws_size,
                              hipStream_t stream) {
    const float* reps   = (const float*)d_in[0];
    const int*   labels = (const int*)d_in[1];
    float*       out    = (float*)d_out;

    char* ws = (char*)d_ws;
    _Float16* xnh  = (_Float16*)ws;                                   // N*D f16 = 2 MB
    float* fbase   = (float*)(ws + (size_t)NROWS * DDIM * sizeof(_Float16));
    float* pos_sum = fbase;                 // N
    float* neg_sum = fbase + NROWS;         // N
    float* cnt_sum = fbase + 2 * NROWS;     // N
    float* maxf    = fbase + 3 * NROWS;     // 1 (uint-bits of positive float)
    float* acc     = fbase + 3 * NROWS + 1; // 2: [masked loss sum, mask count]

    const int nzero = 3 * NROWS + 3;
    zero_f32<<<(nzero + 255) / 256, 256, 0, stream>>>(fbase, nzero);

    normalize_rows<<<NROWS, DDIM, 0, stream>>>(reps, xnh);

    dim3 grid(ROW_TILES / WAVES_PER_BLOCK, JCHUNKS);  // 64 x 16 blocks, 8 waves each
    pass_max <<<grid, 32 * WAVES_PER_BLOCK, 0, stream>>>(xnh, maxf);
    pass_sums<<<grid, 32 * WAVES_PER_BLOCK, 0, stream>>>(xnh, labels, maxf,
                                                         pos_sum, neg_sum, cnt_sum);

    finalize_rows<<<NROWS / 256, 256, 0, stream>>>(pos_sum, neg_sum, cnt_sum, acc);
    write_out<<<1, 1, 0, stream>>>(acc, out);
}
